// ParallelMultiHeadAttention_33595234189827
// MI455X (gfx1250) — matrix operand order repaired and checked by reading, not compiled, not run on hardware
//
#include <hip/hip_runtime.h>
#include <hip/hip_bf16.h>

typedef __attribute__((ext_vector_type(16))) _Float16 v16h;
typedef __attribute__((ext_vector_type(8)))  _Float16 v8h;
typedef __attribute__((ext_vector_type(8)))  float    v8f;

union V16 { v16h v; v8h h[2]; };

#define EMBED 1024
#define HEADS 16
#define HSIZE 64
#define SEQ   2048
#define BATCH 4
#define QKV_N 3072

// ---------------- f32 -> f16 convert ----------------
__global__ void cvt_f32_to_f16(const float* __restrict__ src,
                               _Float16* __restrict__ dst, int n) {
  int i = blockIdx.x * blockDim.x + threadIdx.x;
  int stride = gridDim.x * blockDim.x;
  for (; i < n; i += stride) dst[i] = (_Float16)src[i];
}

// ---------------- NT GEMM core: C[m,n] = sum_k A[m,k]*B[n,k] ----------------
// 8 waves/block, wave grid 4(M)x2(N); each wave 32x64 (8 WMMA tiles);
// block tile 128x128. Register double-buffered; B pointers kept as scalar
// named pointers so loads stay global_load_b128 (not flat_).
template <bool F16OUT>
__device__ __forceinline__ void gemm_nt_body(const _Float16* __restrict__ A,
                                             const _Float16* __restrict__ B,
                                             const float* __restrict__ bias,
                                             void* __restrict__ Cout,
                                             int M, int N, int K) {
  const int lane = threadIdx.x & 31;
  const int wave = threadIdx.x >> 5;
  const int wm = wave & 3, wn = wave >> 2;
  const int mbase = blockIdx.y * 128 + wm * 32;
  const int nbase = blockIdx.x * 128 + wn * 64;
  const int ln = lane & 15;
  const int hi = lane >> 4;

  v8f acc[2][4] = {};
  const _Float16* pa0 = A + (size_t)(mbase + ln) * K + hi * 16;
  const _Float16* pa1 = A + (size_t)(mbase + 16 + ln) * K + hi * 16;
  const _Float16* pb0 = B + (size_t)(nbase + 0 * 16 + ln) * K + hi * 16;
  const _Float16* pb1 = B + (size_t)(nbase + 1 * 16 + ln) * K + hi * 16;
  const _Float16* pb2 = B + (size_t)(nbase + 2 * 16 + ln) * K + hi * 16;
  const _Float16* pb3 = B + (size_t)(nbase + 3 * 16 + ln) * K + hi * 16;

  V16 a0c, a1c, b0c, b1c, b2c, b3c;
  // prologue: load chunk 0
  a0c.h[0] = *(const v8h*)(pa0);      a0c.h[1] = *(const v8h*)(pa0 + 8);
  a1c.h[0] = *(const v8h*)(pa1);      a1c.h[1] = *(const v8h*)(pa1 + 8);
  b0c.h[0] = *(const v8h*)(pb0);      b0c.h[1] = *(const v8h*)(pb0 + 8);
  b1c.h[0] = *(const v8h*)(pb1);      b1c.h[1] = *(const v8h*)(pb1 + 8);
  b2c.h[0] = *(const v8h*)(pb2);      b2c.h[1] = *(const v8h*)(pb2 + 8);
  b3c.h[0] = *(const v8h*)(pb3);      b3c.h[1] = *(const v8h*)(pb3 + 8);

  for (int k = 0; k + 32 < K; k += 32) {
    pa0 += 32; pa1 += 32; pb0 += 32; pb1 += 32; pb2 += 32; pb3 += 32;
    if (k + 160 < K) {
      __builtin_prefetch(pa0 + 128, 0, 1);
      __builtin_prefetch(pa1 + 128, 0, 1);
      __builtin_prefetch(pb0 + 128, 0, 1);
      __builtin_prefetch(pb1 + 128, 0, 1);
      __builtin_prefetch(pb2 + 128, 0, 1);
      __builtin_prefetch(pb3 + 128, 0, 1);
    }
    // issue next chunk's loads first...
    V16 a0n, a1n, b0n, b1n, b2n, b3n;
    a0n.h[0] = *(const v8h*)(pa0);    a0n.h[1] = *(const v8h*)(pa0 + 8);
    a1n.h[0] = *(const v8h*)(pa1);    a1n.h[1] = *(const v8h*)(pa1 + 8);
    b0n.h[0] = *(const v8h*)(pb0);    b0n.h[1] = *(const v8h*)(pb0 + 8);
    b1n.h[0] = *(const v8h*)(pb1);    b1n.h[1] = *(const v8h*)(pb1 + 8);
    b2n.h[0] = *(const v8h*)(pb2);    b2n.h[1] = *(const v8h*)(pb2 + 8);
    b3n.h[0] = *(const v8h*)(pb3);    b3n.h[1] = *(const v8h*)(pb3 + 8);
    // ...then consume current chunk
    acc[0][0] = __builtin_amdgcn_wmma_f32_16x16x32_f16(false, a0c.v, false, b0c.v, (short)0, acc[0][0], false, false);
    acc[1][0] = __builtin_amdgcn_wmma_f32_16x16x32_f16(false, a1c.v, false, b0c.v, (short)0, acc[1][0], false, false);
    acc[0][1] = __builtin_amdgcn_wmma_f32_16x16x32_f16(false, a0c.v, false, b1c.v, (short)0, acc[0][1], false, false);
    acc[1][1] = __builtin_amdgcn_wmma_f32_16x16x32_f16(false, a1c.v, false, b1c.v, (short)0, acc[1][1], false, false);
    acc[0][2] = __builtin_amdgcn_wmma_f32_16x16x32_f16(false, a0c.v, false, b2c.v, (short)0, acc[0][2], false, false);
    acc[1][2] = __builtin_amdgcn_wmma_f32_16x16x32_f16(false, a1c.v, false, b2c.v, (short)0, acc[1][2], false, false);
    acc[0][3] = __builtin_amdgcn_wmma_f32_16x16x32_f16(false, a0c.v, false, b3c.v, (short)0, acc[0][3], false, false);
    acc[1][3] = __builtin_amdgcn_wmma_f32_16x16x32_f16(false, a1c.v, false, b3c.v, (short)0, acc[1][3], false, false);
    a0c = a0n; a1c = a1n; b0c = b0n; b1c = b1n; b2c = b2n; b3c = b3n;
  }
  // epilogue: last chunk
  acc[0][0] = __builtin_amdgcn_wmma_f32_16x16x32_f16(false, a0c.v, false, b0c.v, (short)0, acc[0][0], false, false);
  acc[1][0] = __builtin_amdgcn_wmma_f32_16x16x32_f16(false, a1c.v, false, b0c.v, (short)0, acc[1][0], false, false);
  acc[0][1] = __builtin_amdgcn_wmma_f32_16x16x32_f16(false, a0c.v, false, b1c.v, (short)0, acc[0][1], false, false);
  acc[1][1] = __builtin_amdgcn_wmma_f32_16x16x32_f16(false, a1c.v, false, b1c.v, (short)0, acc[1][1], false, false);
  acc[0][2] = __builtin_amdgcn_wmma_f32_16x16x32_f16(false, a0c.v, false, b2c.v, (short)0, acc[0][2], false, false);
  acc[1][2] = __builtin_amdgcn_wmma_f32_16x16x32_f16(false, a1c.v, false, b2c.v, (short)0, acc[1][2], false, false);
  acc[0][3] = __builtin_amdgcn_wmma_f32_16x16x32_f16(false, a0c.v, false, b3c.v, (short)0, acc[0][3], false, false);
  acc[1][3] = __builtin_amdgcn_wmma_f32_16x16x32_f16(false, a1c.v, false, b3c.v, (short)0, acc[1][3], false, false);

#pragma unroll
  for (int mt = 0; mt < 2; ++mt) {
#pragma unroll
    for (int nt = 0; nt < 4; ++nt) {
      int col = nbase + nt * 16 + ln;
#pragma unroll
      for (int r = 0; r < 8; ++r) {
        int row = mbase + mt * 16 + r + hi * 8;
        if (F16OUT) {
          ((_Float16*)Cout)[(size_t)row * N + col] = (_Float16)acc[mt][nt][r];
        } else {
          ((float*)Cout)[(size_t)row * N + col] = acc[mt][nt][r] + bias[col];
        }
      }
    }
  }
}

__global__ __launch_bounds__(256)
void gemm_nt_f16out(const _Float16* __restrict__ A, const _Float16* __restrict__ B,
                    _Float16* __restrict__ C, int M, int N, int K) {
  gemm_nt_body<true>(A, B, nullptr, C, M, N, K);
}

__global__ __launch_bounds__(256)
void gemm_nt_f32out_bias(const _Float16* __restrict__ A, const _Float16* __restrict__ B,
                         const float* __restrict__ bias, float* __restrict__ C,
                         int M, int N, int K) {
  gemm_nt_body<false>(A, B, bias, C, M, N, K);
}

// ---------------- Flash attention ----------------
// 4 waves/block over 64 consecutive queries of one (b,h). K/V 32-key chunks
// are cooperatively staged into ping-pong LDS buffers: chunk kc+1 is staged
// while chunk kc is computed; one barrier per chunk.
__global__ __launch_bounds__(128)
void flash_attn(const _Float16* __restrict__ qkv, _Float16* __restrict__ out) {
  __shared__ __align__(32) _Float16 lds_k[2][32 * 64];     // [buf][key][d]
  __shared__ __align__(32) _Float16 lds_vt[2][64 * 32];    // [buf][d][key]
  __shared__ __align__(32) _Float16 lds_p[4 * 16 * 32];    // per-wave P tile

  const int tid  = threadIdx.x;
  const int lane = tid & 31;
  const int wave = tid >> 5;
  const int ln = lane & 15;
  const int hi = lane >> 4;

  const int id  = blockIdx.x;
  const int qt0 = id & 31;            // 64-query group (SEQ/64 = 32)
  const int h   = (id >> 5) & 15;
  const int b   = id >> 9;

  const size_t rs3 = (size_t)QKV_N;
  const _Float16* base = qkv + (size_t)b * SEQ * rs3;

  const int my_q0   = qt0 * 64 + wave * 16;
  const int my_qmax = my_q0 + 15;

  const int sr  = tid >> 2;
  const int seg = (tid & 3) * 16;
  const _Float16* krow_g = base + (size_t)sr * rs3 + EMBED + h * HSIZE + seg;
  const _Float16* vrow_g = base + (size_t)sr * rs3 + 2 * EMBED + h * HSIZE + seg;
  const size_t chunk_step = 32 * rs3;

  auto stage = [&](int kc, int buf) {
    const _Float16* kg = krow_g + (size_t)kc * chunk_step;
    v8h k0 = *(const v8h*)(kg);
    v8h k1 = *(const v8h*)(kg + 8);
    *(v8h*)(&lds_k[buf][sr * 64 + seg])     = k0;
    *(v8h*)(&lds_k[buf][sr * 64 + seg + 8]) = k1;
    const _Float16* vg = vrow_g + (size_t)kc * chunk_step;
    v8h v0 = *(const v8h*)(vg);
    v8h v1 = *(const v8h*)(vg + 8);
#pragma unroll
    for (int e = 0; e < 8; ++e) {
      lds_vt[buf][(seg + e) * 32 + sr]     = v0[e];
      lds_vt[buf][(seg + 8 + e) * 32 + sr] = v1[e];
    }
  };

  // Q fragment (A layout), two d-chunks of 32
  V16 aQ[2];
  {
    const _Float16* qrow = base + (size_t)(my_q0 + ln) * rs3 + h * HSIZE;
#pragma unroll
    for (int c = 0; c < 2; ++c) {
      int g0 = c * 32 + hi * 16;
      aQ[c].h[0] = *(const v8h*)(qrow + g0);
      aQ[c].h[1] = *(const v8h*)(qrow + g0 + 8);
    }
  }

  float mrow[8], lrow[8];
  v8f o[4] = {};
#pragma unroll
  for (int r = 0; r < 8; ++r) { mrow[r] = -1e30f; lrow[r] = 0.f; }

  _Float16* pmine = lds_p + wave * (16 * 32);
  const int nchunks = (qt0 * 64 + 63) / 32 + 1;

  stage(0, 0);

  for (int kc = 0; kc < nchunks; ++kc) {
    const int kb = kc * 32;
    const int cur = kc & 1;
    __syncthreads();                      // staging of buf `cur` complete
    if (kc + 1 < nchunks) stage(kc + 1, cur ^ 1);

    if (kb <= my_qmax) {   // wave-uniform: EXEC all-ones inside
      const _Float16* Lk = &lds_k[cur][0];
      const _Float16* Lv = &lds_vt[cur][0];

      // ---- S = Q K^T ----
      v8f s[2];
#pragma unroll
      for (int t = 0; t < 2; ++t) {
        const _Float16* krow = Lk + (t * 16 + ln) * 64;
        v8f z = {};
#pragma unroll
        for (int c = 0; c < 2; ++c) {
          V16 bk;
          int b0 = c * 32 + hi * 16;
          bk.h[0] = *(const v8h*)(krow + b0);
          bk.h[1] = *(const v8h*)(krow + b0 + 8);
          z = __builtin_amdgcn_wmma_f32_16x16x32_f16(
              false, aQ[c].v, false, bk.v, (short)0, z, false, false);
        }
        s[t] = z;
      }

      // ---- causal mask + scale ----
#pragma unroll
      for (int t = 0; t < 2; ++t) {
        int key = kb + t * 16 + ln;
#pragma unroll
        for (int r = 0; r < 8; ++r) {
          int q = my_q0 + r + hi * 8;
          float v = s[t][r] * 0.125f;
          s[t][r] = (key > q) ? -1e30f : v;
        }
      }

      // ---- online softmax ----
      float p0[8], p1[8];
#pragma unroll
      for (int r = 0; r < 8; ++r) {
        float cm = fmaxf(s[0][r], s[1][r]);
        cm = fmaxf(cm, __shfl_xor(cm, 1, 32));
        cm = fmaxf(cm, __shfl_xor(cm, 2, 32));
        cm = fmaxf(cm, __shfl_xor(cm, 4, 32));
        cm = fmaxf(cm, __shfl_xor(cm, 8, 32));
        float mnew = fmaxf(mrow[r], cm);
        float alpha = __expf(mrow[r] - mnew);
        mrow[r] = mnew;
        p0[r] = __expf(s[0][r] - mnew);
        p1[r] = __expf(s[1][r] - mnew);
        float rsum = p0[r] + p1[r];
        rsum += __shfl_xor(rsum, 1, 32);
        rsum += __shfl_xor(rsum, 2, 32);
        rsum += __shfl_xor(rsum, 4, 32);
        rsum += __shfl_xor(rsum, 8, 32);
        lrow[r] = lrow[r] * alpha + rsum;
#pragma unroll
        for (int c = 0; c < 4; ++c) o[c][r] *= alpha;
      }

      // ---- P (C layout) -> per-wave LDS -> A fragment ----
#pragma unroll
      for (int r = 0; r < 8; ++r) {
        int m = r + hi * 8;
        pmine[m * 32 + ln]      = (_Float16)p0[r];
        pmine[m * 32 + 16 + ln] = (_Float16)p1[r];
      }
      V16 ap;
      {
        const _Float16* prow = pmine + ln * 32;
        int g0 = hi * 16;
        ap.h[0] = *(const v8h*)(prow + g0);
        ap.h[1] = *(const v8h*)(prow + g0 + 8);
      }

      // ---- O += P V ----
#pragma unroll
      for (int c = 0; c < 4; ++c) {
        V16 bv;
        const _Float16* vcol = Lv + (c * 16 + ln) * 32 + hi * 16;
        bv.h[0] = *(const v8h*)(vcol);
        bv.h[1] = *(const v8h*)(vcol + 8);
        o[c] = __builtin_amdgcn_wmma_f32_16x16x32_f16(
            false, ap.v, false, bv.v, (short)0, o[c], false, false);
      }
    }
  }

  // ---- normalize + store ----
#pragma unroll
  for (int r = 0; r < 8; ++r) {
    float inv = 1.0f / lrow[r];
    int q = my_q0 + r + hi * 8;
    _Float16* orow = out + (size_t)(b * SEQ + q) * EMBED + h * HSIZE;
#pragma unroll
    for (int c = 0; c < 4; ++c) orow[c * 16 + ln] = (_Float16)(o[c][r] * inv);
  }
}

extern "C" void kernel_launch(void* const* d_in, const int* in_sizes, int n_in,
                              void* d_out, int out_size, void* d_ws, size_t ws_size,
                              hipStream_t stream) {
  const float* x      = (const float*)d_in[0];
  const float* w_qkv  = (const float*)d_in[1];
  const float* w_proj = (const float*)d_in[2];
  const float* b_proj = (const float*)d_in[3];
  float* out = (float*)d_out;

  const int M = BATCH * SEQ;               // 8192
  _Float16* x_h     = (_Float16*)d_ws;                       // M*EMBED
  _Float16* wqkv_h  = x_h + (size_t)M * EMBED;               // QKV_N*EMBED
  _Float16* wproj_h = wqkv_h + (size_t)QKV_N * EMBED;        // EMBED*EMBED
  _Float16* qkv_h   = wproj_h + (size_t)EMBED * EMBED;       // M*QKV_N
  _Float16* attn_h  = qkv_h + (size_t)M * QKV_N;             // M*EMBED

  cvt_f32_to_f16<<<4096, 256, 0, stream>>>(x, x_h, M * EMBED);
  cvt_f32_to_f16<<<2048, 256, 0, stream>>>(w_qkv, wqkv_h, QKV_N * EMBED);
  cvt_f32_to_f16<<<1024, 256, 0, stream>>>(w_proj, wproj_h, EMBED * EMBED);

  // QKV projection: [8192,1024] x [3072,1024]^T -> [8192,3072] f16
  gemm_nt_f16out<<<dim3(QKV_N / 128, M / 128), 256, 0, stream>>>(
      x_h, wqkv_h, qkv_h, M, QKV_N, EMBED);

  // Attention: 4 waves / block, 64 queries per block
  flash_attn<<<BATCH * HEADS * (SEQ / 64), 128, 0, stream>>>(qkv_h, attn_h);

  // Output projection with bias: [8192,1024] x [1024,1024]^T + b -> f32
  gemm_nt_f32out_bias<<<dim3(EMBED / 128, M / 128), 256, 0, stream>>>(
      attn_h, wproj_h, b_proj, out, M, EMBED, EMBED);
}
